// MultiHeadSelfAttention_24893630447824
// MI455X (gfx1250) — matrix operand order repaired and checked by reading, not compiled, not run on hardware
//
#include <hip/hip_runtime.h>
#include <hip/hip_bf16.h>
#include <stdint.h>

#define B_   4
#define T_   2048
#define C_   1024
#define H_   16
#define DH_  64
#define DFF_ 4096
#define BT_  (B_*T_)

typedef __attribute__((ext_vector_type(16))) __bf16 v16bf;
typedef __attribute__((ext_vector_type(8)))  float  v8f;

union Frag { v16bf b; unsigned int u[8]; };

__device__ __forceinline__ unsigned short f2bf(float f) {
  unsigned u = __float_as_uint(f);
  u += 0x7FFFu + ((u >> 16) & 1u);           // round-to-nearest-even
  return (unsigned short)(u >> 16);
}
__device__ __forceinline__ unsigned int pack2(unsigned short lo, unsigned short hi) {
  return (unsigned int)lo | ((unsigned int)hi << 16);
}

// ---------------------------------------------------------------- pack kernels
// Q weights/bias are pre-scaled by sqrt(DH)=8 (faithful to reference's scores*sqrt(DH)),
// so the attention kernel needs no per-score scaling.
__global__ void pack_qkv_kernel(const float* __restrict__ Wq, const float* __restrict__ Wk,
                                const float* __restrict__ Wv, const float* __restrict__ bq,
                                const float* __restrict__ bk, const float* __restrict__ bv,
                                unsigned short* __restrict__ Wqkv, float* __restrict__ bias_qkv) {
  int idx = blockIdx.x * blockDim.x + threadIdx.x;
  if (idx < 3*C_) {
    int j = idx;
    bias_qkv[j] = (j < C_) ? 8.0f * bq[j] : (j < 2*C_) ? bk[j - C_] : bv[j - 2*C_];
  }
  if (idx >= C_*3*C_) return;
  int c   = idx / (3*C_);
  int j   = idx % (3*C_);
  int sel = j / C_;
  int jj  = j % C_;
  int h = jj >> 6, d = jj & 63;
  const float* W = (sel == 0) ? Wq : (sel == 1) ? Wk : Wv;
  float w = W[((size_t)h * C_ + c) * DH_ + d];
  if (sel == 0) w *= 8.0f;
  Wqkv[(size_t)c * (3*C_) + j] = f2bf(w);
}

__global__ void pack_bf16_kernel(const float* __restrict__ src,
                                 unsigned short* __restrict__ dst, int n) {
  int i = blockIdx.x * blockDim.x + threadIdx.x;
  if (i < n) dst[i] = f2bf(src[i]);
}

// ---------------------------------------------------------------- layernorm
__global__ __launch_bounds__(256) void layernorm_bf16_kernel(
    const float* __restrict__ x, const float* __restrict__ g,
    const float* __restrict__ b, unsigned short* __restrict__ out) {
  __shared__ float red[16];
  int row = blockIdx.x;
  int t = threadIdx.x;
  const float* xr = x + (size_t)row * C_;
  float v[4], s = 0.f, s2 = 0.f;
  #pragma unroll
  for (int i = 0; i < 4; i++) { v[i] = xr[t + i*256]; s += v[i]; s2 += v[i]*v[i]; }
  #pragma unroll
  for (int off = 16; off >= 1; off >>= 1) {
    s  += __shfl_xor(s,  off, 32);
    s2 += __shfl_xor(s2, off, 32);
  }
  int wid = t >> 5, lane = t & 31;
  if (lane == 0) { red[wid] = s; red[8 + wid] = s2; }
  __syncthreads();
  if (t < 32) {
    float a  = (t < 8) ? red[t]     : 0.f;
    float a2 = (t < 8) ? red[8 + t] : 0.f;
    #pragma unroll
    for (int off = 4; off >= 1; off >>= 1) {
      a  += __shfl_xor(a,  off, 32);
      a2 += __shfl_xor(a2, off, 32);
    }
    if (t == 0) { red[0] = a; red[1] = a2; }
  }
  __syncthreads();
  float mu  = red[0] * (1.f / C_);
  float var = red[1] * (1.f / C_) - mu * mu;
  float inv = rsqrtf(var + 1e-5f);
  unsigned short* orow = out + (size_t)row * C_;
  #pragma unroll
  for (int i = 0; i < 4; i++) {
    int c = t + i*256;
    orow[c] = f2bf((v[i] - mu) * inv * g[c] + b[c]);
  }
}

// ---------------------------------------------------------------- GEMM (bf16 in, f32 acc)
// Block: 256 thr = 8 waves (2x4). Wave tile 32x64 (2x4 WMMA accums).
// Block tile 64x256, K-step 32. Requires M%64==0, N%256==0, K%32==0.
template<bool RELU, bool RES, bool BF16OUT>
__global__ __launch_bounds__(256) void gemm_bf16_kernel(
    const unsigned short* __restrict__ A, const unsigned short* __restrict__ Bm,
    const float* __restrict__ bias, const float* __restrict__ resid,
    float* __restrict__ outF, unsigned short* __restrict__ outB,
    int M, int N, int K) {
  __shared__ unsigned short As[64 * 40];    // [row][k]  pad 40
  __shared__ unsigned short Bs[256 * 40];   // [col][k]  (transposed) pad 40
  const int t = threadIdx.x;
  const int lane = t & 31, wid = t >> 5;
  const int l15 = lane & 15, lh = lane >> 4;
  const int wm = wid >> 2, wn = wid & 3;
  const int tileM = blockIdx.y * 64;
  const int tileN = blockIdx.x * 256;

  v8f acc[2][4] = {};

  const int arow = t >> 2, acol = (t & 3) * 8;   // A stage: 64x32, 8 elems/thr
  const int brp  = t & 15, bcs  = (t >> 4) * 16; // B stage: k-row pair x 16 cols

  int k0a[8], k0b[8];
  #pragma unroll
  for (int i = 0; i < 8; i++) {
    k0a[i] = 2*i + (i >= 4 ? 8 : 0) + 8*lh;   // A-frag K pairs (ISA 16-bit A 16x32)
    k0b[i] = 2*i + (i >= 4 ? 8 : 0) + 8*lh;   // B-frag K pairs (same K order as A)
  }

  for (int kt = 0; kt < K; kt += 32) {
    uint4 av = *(const uint4*)(A + (size_t)(tileM + arow) * K + kt + acol);
    union { uint4 q[2]; unsigned short h[16]; } r0, r1;
    const unsigned short* bp0 = Bm + (size_t)(kt + 2*brp)     * N + tileN + bcs;
    const unsigned short* bp1 = Bm + (size_t)(kt + 2*brp + 1) * N + tileN + bcs;
    r0.q[0] = *(const uint4*)bp0;  r0.q[1] = *(const uint4*)(bp0 + 8);
    r1.q[0] = *(const uint4*)bp1;  r1.q[1] = *(const uint4*)(bp1 + 8);
    __builtin_prefetch(A + (size_t)(tileM + arow) * K + kt + 32 + acol, 0, 1);
    __builtin_prefetch(bp0 + (size_t)32 * N, 0, 1);
    __builtin_prefetch(bp1 + (size_t)32 * N, 0, 1);
    __syncthreads();                           // prior tile fully consumed
    *(uint4*)&As[arow * 40 + acol] = av;
    #pragma unroll
    for (int e = 0; e < 16; e++)
      *(unsigned int*)&Bs[(bcs + e) * 40 + 2*brp] = pack2(r0.h[e], r1.h[e]);
    __syncthreads();

    Frag af[2], bfr[4];
    #pragma unroll
    for (int mi = 0; mi < 2; mi++) {
      int r = wm*32 + mi*16 + l15;
      #pragma unroll
      for (int i = 0; i < 8; i++) af[mi].u[i] = *(const unsigned int*)&As[r * 40 + k0a[i]];
    }
    #pragma unroll
    for (int ni = 0; ni < 4; ni++) {
      int cc = wn*64 + ni*16 + l15;
      #pragma unroll
      for (int i = 0; i < 8; i++) bfr[ni].u[i] = *(const unsigned int*)&Bs[cc * 40 + k0b[i]];
    }
    #pragma unroll
    for (int mi = 0; mi < 2; mi++)
      #pragma unroll
      for (int ni = 0; ni < 4; ni++)
        acc[mi][ni] = __builtin_amdgcn_wmma_f32_16x16x32_bf16(
            false, af[mi].b, false, bfr[ni].b, (short)0, acc[mi][ni], false, false);
  }

  #pragma unroll
  for (int ni = 0; ni < 4; ni++) {
    int col = tileN + wn*64 + ni*16 + l15;
    float bcol = bias[col];
    #pragma unroll
    for (int mi = 0; mi < 2; mi++)
      #pragma unroll
      for (int v = 0; v < 8; v++) {
        int row = tileM + wm*32 + mi*16 + v + 8*lh;
        float val = acc[mi][ni][v] + bcol;
        if (RELU) val = fmaxf(val, 0.f);
        size_t o = (size_t)row * N + col;
        if (RES) val += resid[o];
        if (BF16OUT) outB[o] = f2bf(val);
        else         outF[o] = val;
      }
  }
}

// ---------------------------------------------------------------- flash attention
// qkv: [BT][3C] bf16 (Q pre-scaled by 8). One block = one (b,h) x 128 query rows;
// each wave owns 16 rows. Online softmax over 64-key chunks; row-sum accumulated
// by an extra WMMA against an all-ones B matrix (no shuffle tree for sums).
__global__ __launch_bounds__(256) void attn_kernel(const unsigned short* __restrict__ qkv,
                                                   unsigned short* __restrict__ o_ws) {
  __shared__ unsigned short Ks[64 * 64];     // [s][c] row-major
  __shared__ unsigned short Vs[64 * 72];     // [d][s] transposed, pad 72
  __shared__ unsigned short Ps[8][16 * 72];  // per-wave P (16 rows x 64 keys), pad 72
  const int t = threadIdx.x;
  const int lane = t & 31, wid = t >> 5;
  const int l15 = lane & 15, lh = lane >> 4;
  const int bh = blockIdx.y;
  const int b = bh >> 4, h = bh & 15;
  const int qbase = blockIdx.x * 128 + wid * 16;
  const size_t rs3 = (size_t)(3 * C_);
  const unsigned short* Qp = qkv + (size_t)(b * T_) * rs3 + h * DH_;
  const unsigned short* Kp = Qp + C_;
  const unsigned short* Vp = Qp + 2 * C_;

  int k0a[8], k0b[8];
  #pragma unroll
  for (int i = 0; i < 8; i++) {
    k0a[i] = 2*i + (i >= 4 ? 8 : 0) + 8*lh;
    k0b[i] = 2*i + (i >= 4 ? 8 : 0) + 8*lh;
  }

  Frag qf[2];                                // resident Q A-frags (16x64)
  #pragma unroll
  for (int ck = 0; ck < 2; ck++)
    #pragma unroll
    for (int i = 0; i < 8; i++)
      qf[ck].u[i] = *(const unsigned int*)(Qp + (size_t)(qbase + l15) * rs3 + ck*32 + k0a[i]);

  Frag ones;                                 // bf16 1.0 everywhere (B = ones 32x16)
  #pragma unroll
  for (int i = 0; i < 8; i++) ones.u[i] = 0x3F803F80u;

  float m[8];
  #pragma unroll
  for (int v = 0; v < 8; v++) m[v] = -3.0e38f;
  v8f oacc[4] = {};
  v8f lacc = {};

  const int krow = t >> 2, kseg = (t & 3) * 16;  // K staging: 64x64, 16 elems/thr
  const int vrp  = t & 31, vds  = (t >> 5) * 8;  // V staging: s-row pair x 8 d-cols

  for (int s0 = 0; s0 < T_; s0 += 64) {
    uint4 ka = *(const uint4*)(Kp + (size_t)(s0 + krow) * rs3 + kseg);
    uint4 kb = *(const uint4*)(Kp + (size_t)(s0 + krow) * rs3 + kseg + 8);
    union { uint4 q; unsigned short h[8]; } v0, v1;
    v0.q = *(const uint4*)(Vp + (size_t)(s0 + 2*vrp)     * rs3 + vds);
    v1.q = *(const uint4*)(Vp + (size_t)(s0 + 2*vrp + 1) * rs3 + vds);
    __syncthreads();
    *(uint4*)&Ks[krow * 64 + kseg]     = ka;
    *(uint4*)&Ks[krow * 64 + kseg + 8] = kb;
    #pragma unroll
    for (int e = 0; e < 8; e++)
      *(unsigned int*)&Vs[(vds + e) * 72 + 2*vrp] = pack2(v0.h[e], v1.h[e]);
    __syncthreads();

    // S = Q*K^T for 64 keys (four 16x16 tiles, 2 WMMAs each over DH=64)
    v8f sacc[4] = {};
    #pragma unroll
    for (int ssi = 0; ssi < 4; ssi++)
      #pragma unroll
      for (int ck = 0; ck < 2; ck++) {
        Frag kf;
        #pragma unroll
        for (int i = 0; i < 8; i++)
          kf.u[i] = *(const unsigned int*)&Ks[(ssi*16 + l15) * 64 + ck*32 + k0b[i]];
        sacc[ssi] = __builtin_amdgcn_wmma_f32_16x16x32_bf16(
            false, qf[ck].b, false, kf.b, (short)0, sacc[ssi], false, false);
      }

    // row max over 64 keys: 3 VALU maxes + one 4-step shuffle tree per row
    float nmv[8];
    #pragma unroll
    for (int v = 0; v < 8; v++) {
      float r = fmaxf(fmaxf(sacc[0][v], sacc[1][v]), fmaxf(sacc[2][v], sacc[3][v]));
      #pragma unroll
      for (int off = 1; off < 16; off <<= 1) r = fmaxf(r, __shfl_xor(r, off, 32));
      float nm = fmaxf(m[v], r);
      float alpha = __expf(m[v] - nm);
      m[v] = nm;
      nmv[v] = nm;
      lacc[v] *= alpha;
      #pragma unroll
      for (int dg = 0; dg < 4; dg++) oacc[dg][v] *= alpha;
    }

    // P = exp(S - m), written straight into A-frag repack buffer
    unsigned short* pw = &Ps[wid][0];
    #pragma unroll
    for (int ssi = 0; ssi < 4; ssi++)
      #pragma unroll
      for (int v = 0; v < 8; v++)
        pw[(v + 8*lh) * 72 + ssi*16 + l15] = f2bf(__expf(sacc[ssi][v] - nmv[v]));

    Frag pf[2];
    #pragma unroll
    for (int ck = 0; ck < 2; ck++)
      #pragma unroll
      for (int i = 0; i < 8; i++)
        pf[ck].u[i] = *(const unsigned int*)&pw[l15 * 72 + ck*32 + k0a[i]];

    // l += P * ones  (row sums via matrix core)
    #pragma unroll
    for (int ck = 0; ck < 2; ck++)
      lacc = __builtin_amdgcn_wmma_f32_16x16x32_bf16(
          false, pf[ck].b, false, ones.b, (short)0, lacc, false, false);

    // O += P * V  (4 d-groups x 2 k-chunks)
    #pragma unroll
    for (int dg = 0; dg < 4; dg++)
      #pragma unroll
      for (int ck = 0; ck < 2; ck++) {
        Frag vf;
        #pragma unroll
        for (int i = 0; i < 8; i++)
          vf.u[i] = *(const unsigned int*)&Vs[(dg*16 + l15) * 72 + ck*32 + k0b[i]];
        oacc[dg] = __builtin_amdgcn_wmma_f32_16x16x32_bf16(
            false, pf[ck].b, false, vf.b, (short)0, oacc[dg], false, false);
      }
  }

  #pragma unroll
  for (int dg = 0; dg < 4; dg++)
    #pragma unroll
    for (int v = 0; v < 8; v++) {
      int row = qbase + v + 8*lh;
      int col = h * DH_ + dg*16 + l15;
      o_ws[(size_t)(b * T_ + row) * C_ + col] = f2bf(oacc[dg][v] / lacc[v]);
    }
}

// ---------------------------------------------------------------- launch
extern "C" void kernel_launch(void* const* d_in, const int* in_sizes, int n_in,
                              void* d_out, int out_size, void* d_ws, size_t ws_size,
                              hipStream_t stream) {
  (void)in_sizes; (void)n_in; (void)out_size; (void)ws_size;
  const float* x    = (const float*)d_in[0];
  const float* Wq   = (const float*)d_in[1];
  const float* bq   = (const float*)d_in[2];
  const float* Wk   = (const float*)d_in[3];
  const float* bk   = (const float*)d_in[4];
  const float* Wv   = (const float*)d_in[5];
  const float* bv   = (const float*)d_in[6];
  const float* Wo   = (const float*)d_in[7];
  const float* bo   = (const float*)d_in[8];
  const float* ln1g = (const float*)d_in[9];
  const float* ln1b = (const float*)d_in[10];
  const float* ln2g = (const float*)d_in[11];
  const float* ln2b = (const float*)d_in[12];
  const float* W1   = (const float*)d_in[13];
  const float* b1   = (const float*)d_in[14];
  const float* W2   = (const float*)d_in[15];
  const float* b2   = (const float*)d_in[16];
  float* out = (float*)d_out;

  char* p = (char*)d_ws;
  auto carve = [&](size_t bytes) -> char* {
    char* r = p; p += (bytes + 255) & ~(size_t)255; return r;
  };
  unsigned short* xn    = (unsigned short*)carve((size_t)BT_ * C_ * 2);
  unsigned short* qkv   = (unsigned short*)carve((size_t)BT_ * 3 * C_ * 2);
  unsigned short* ows   = (unsigned short*)carve((size_t)BT_ * C_ * 2);
  float*          x1    = (float*)carve((size_t)BT_ * C_ * 4);
  unsigned short* xn2   = (unsigned short*)carve((size_t)BT_ * C_ * 2);
  unsigned short* h1    = (unsigned short*)carve((size_t)BT_ * DFF_ * 2);
  unsigned short* Wqkvb = (unsigned short*)carve((size_t)C_ * 3 * C_ * 2);
  float*          bqkv  = (float*)carve((size_t)3 * C_ * 4);
  unsigned short* Wob   = (unsigned short*)carve((size_t)C_ * C_ * 2);
  unsigned short* W1b   = (unsigned short*)carve((size_t)C_ * DFF_ * 2);
  unsigned short* W2b   = (unsigned short*)carve((size_t)DFF_ * C_ * 2);

  pack_qkv_kernel<<<(C_*3*C_ + 255)/256, 256, 0, stream>>>(Wq, Wk, Wv, bq, bk, bv, Wqkvb, bqkv);
  pack_bf16_kernel<<<(C_*C_   + 255)/256, 256, 0, stream>>>(Wo, Wob, C_*C_);
  pack_bf16_kernel<<<(C_*DFF_ + 255)/256, 256, 0, stream>>>(W1, W1b, C_*DFF_);
  pack_bf16_kernel<<<(DFF_*C_ + 255)/256, 256, 0, stream>>>(W2, W2b, DFF_*C_);

  layernorm_bf16_kernel<<<BT_, 256, 0, stream>>>(x, ln1g, ln1b, xn);
  // QKV: bias, no relu, no resid, bf16 out
  gemm_bf16_kernel<false, false, true><<<dim3(3*C_/256, BT_/64), 256, 0, stream>>>(
      xn, Wqkvb, bqkv, nullptr, nullptr, qkv, BT_, 3*C_, C_);
  attn_kernel<<<dim3(T_/128, B_*H_), 256, 0, stream>>>(qkv, ows);
  // Wo: bias + residual(x), f32 out
  gemm_bf16_kernel<false, true, false><<<dim3(C_/256, BT_/64), 256, 0, stream>>>(
      ows, Wob, bo, x, x1, nullptr, BT_, C_, C_);
  layernorm_bf16_kernel<<<BT_, 256, 0, stream>>>(x1, ln2g, ln2b, xn2);
  // FFN1: bias + relu, bf16 out
  gemm_bf16_kernel<true, false, true><<<dim3(DFF_/256, BT_/64), 256, 0, stream>>>(
      xn2, W1b, b1, nullptr, nullptr, h1, BT_, DFF_, C_);
  // FFN2: bias + residual(x1), f32 out
  gemm_bf16_kernel<false, true, false><<<dim3(C_/256, BT_/64), 256, 0, stream>>>(
      h1, W2b, b2, x1, out, nullptr, BT_, C_, DFF_);
}
